// CompensatedSparseLinear_53566832116289
// MI455X (gfx1250) — matrix operand order repaired and checked by reading, not compiled, not run on hardware
//
#include <hip/hip_runtime.h>

#define DIN  4096
#define DOUT 4096
#define KT   32
#define LDSP 40     // LDS row stride in halves (80B: conflict-free, 8B-aligned)
#define MT   128    // block tile M
#define NT   64     // block tile N

typedef __attribute__((ext_vector_type(16))) __bf16       v16bf;
typedef __attribute__((ext_vector_type(8)))  float        v8f;
typedef __attribute__((ext_vector_type(4)))  unsigned int v4u;
typedef __attribute__((ext_vector_type(8)))  int          v8i;
typedef __attribute__((ext_vector_type(4)))  int          v4i;

union FragBF { v16bf v; uint2 q[4]; };

// ---- f32 -> bf16 split (round-to-nearest-even) via bit math ----
__device__ __forceinline__ unsigned short f2bf(float f) {
    unsigned u = __builtin_bit_cast(unsigned, f);
    unsigned r = (u + 0x7FFFu + ((u >> 16) & 1u)) >> 16;
    return (unsigned short)r;
}
__device__ __forceinline__ float bf2f(unsigned short b) {
    return __builtin_bit_cast(float, (unsigned)b << 16);
}
__device__ __forceinline__ void split4(const float4& v, uint2& ph, uint2& pl) {
    union { unsigned short h[4]; uint2 u; } a, b;
    float f[4] = {v.x, v.y, v.z, v.w};
#pragma unroll
    for (int j = 0; j < 4; ++j) {
        unsigned short h = f2bf(f[j]);
        a.h[j] = h;
        b.h[j] = f2bf(f[j] - bf2f(h));
    }
    ph = a.u; pl = b.u;
}

// A fragment: 16x32 bf16, M = lane%16; lanes<16 hold K0-7 (v0-3), K16-23 (v4-7);
// lanes>=16 hold K8-15, K24-31.
__device__ __forceinline__ void load_frag_A(const unsigned short* As, int mBase, FragBF& f) {
    const int lane = threadIdx.x & 31;
    const unsigned short* row = As + (mBase + (lane & 15)) * LDSP;
    const int hb = (lane >> 4) << 3;
    f.q[0] = *(const uint2*)(row + hb);
    f.q[1] = *(const uint2*)(row + hb + 4);
    f.q[2] = *(const uint2*)(row + 16 + hb);
    f.q[3] = *(const uint2*)(row + 20 + hb);
}
// B fragment: 32x16 bf16, N = lane%16; lanes<16 hold K0-7, K16-23; lanes>=16 hold K8-15, K24-31.
__device__ __forceinline__ void load_frag_B(const unsigned short* Bs, int nBase, FragBF& f) {
    const int lane = threadIdx.x & 31;
    const unsigned short* row = Bs + (nBase + (lane & 15)) * LDSP;
    const int hb = (lane >> 4) << 3;
    f.q[0] = *(const uint2*)(row + hb);
    f.q[1] = *(const uint2*)(row + hb + 4);
    f.q[2] = *(const uint2*)(row + 16 + hb);
    f.q[3] = *(const uint2*)(row + 20 + hb);
}

__device__ __forceinline__ v8f wmma_bf16(const FragBF& a, const FragBF& b, v8f c) {
    return __builtin_amdgcn_wmma_f32_16x16x32_bf16(
        false, a.v, false, b.v, (short)0, c, false, false);
}

// ---- TDM: issue a 2D bf16 tile load (rows x 32 halves) Global -> LDS, with HW
// padding producing the 40-half LDS row stride (pad 4 DWORDs every 16 DWORDs). ----
__device__ __forceinline__ void tdm_load_tile(unsigned lds_byte_addr, const void* gaddr,
                                              unsigned tensor_rows, unsigned tile_rows) {
    const unsigned long long ga = (unsigned long long)(uintptr_t)gaddr;
    v4u g0;
    g0[0] = 1u;                                            // count=1, user descriptor
    g0[1] = lds_byte_addr;                                 // lds_addr
    g0[2] = (unsigned)(ga & 0xffffffffull);                // global_addr[31:0]
    g0[3] = (unsigned)((ga >> 32) & 0x1ffffffull) | (2u << 30);  // ga[56:32] | type=2
    v8i g1;
    g1[0] = (int)((1u << 16)     // data_size = 2 bytes
                | (1u << 20)     // pad_enable
                | (3u << 22)     // pad_interval: every 16 DWORDs (64B row)
                | (3u << 25));   // pad_amount: 4 DWORDs (16B) -> 40-half stride
    g1[1] = (int)(((unsigned)DIN & 0xffffu) << 16);        // tensor_dim0[15:0]
    g1[2] = (int)((((unsigned)DIN >> 16) & 0xffffu) | ((tensor_rows & 0xffffu) << 16));
    g1[3] = (int)(((tensor_rows >> 16) & 0xffffu) | (32u << 16)); // tile_dim0 = 32
    g1[4] = (int)(tile_rows & 0xffffu);                    // tile_dim1 (tile_dim2=0)
    g1[5] = (int)DIN;                                      // tensor_dim0_stride[31:0]
    g1[6] = 0;
    g1[7] = 0;
    v4i gz = {0, 0, 0, 0};
#if defined(__clang_major__) && (__clang_major__ >= 23)
    v8i gz8 = {0, 0, 0, 0, 0, 0, 0, 0};
    __builtin_amdgcn_tensor_load_to_lds(g0, g1, gz, gz, gz8, 0);
#else
    __builtin_amdgcn_tensor_load_to_lds(g0, g1, gz, gz, 0);
#endif
}

// ---- Kernel 1: copy W into workspace (W_eff base) ----
__global__ __launch_bounds__(256) void copy_w_kernel(const float4* __restrict__ src,
                                                     float4* __restrict__ dst, int n4) {
    int i = blockIdx.x * 256 + threadIdx.x;
    if (i < n4) dst[i] = src[i];
}

// ---- Kernel 2: scatter-add sparse delta into W_eff ----
__global__ __launch_bounds__(256) void scatter_kernel(const float* __restrict__ vals,
                                                      const int* __restrict__ rows,
                                                      const int* __restrict__ cols,
                                                      float* __restrict__ Weff, int nnz) {
    int i = blockIdx.x * 256 + threadIdx.x;
    if (i < nnz)
        atomicAdd(Weff + (size_t)rows[i] * DIN + cols[i], vals[i]);
}

// ---- Kernel 3: split an f32 plane into bf16 hi/lo planes ----
__global__ __launch_bounds__(256) void split_plane_kernel(const float4* __restrict__ src,
                                                          uint2* __restrict__ hi,
                                                          uint2* __restrict__ lo, int n4) {
    int i = blockIdx.x * 256 + threadIdx.x;
    if (i < n4) {
        uint2 ph, pl;
        split4(src[i], ph, pl);
        hi[i] = ph;
        lo[i] = pl;
    }
}

// ---- Kernel 4 (fast path): TDM double-buffered bf16x2 split-precision WMMA GEMM ----
__global__ __launch_bounds__(256) void gemm_tdm_kernel(const unsigned short* __restrict__ xh,
                                                       const unsigned short* __restrict__ xl,
                                                       const unsigned short* __restrict__ wh,
                                                       const unsigned short* __restrict__ wl,
                                                       const float* __restrict__ bias,
                                                       float* __restrict__ out) {
    constexpr int AH_OFF = 0;
    constexpr int AL_OFF = MT * LDSP * 2;            // 10240
    constexpr int BH_OFF = 2 * MT * LDSP * 2;        // 20480
    constexpr int BL_OFF = BH_OFF + NT * LDSP * 2;   // 25600
    constexpr int BUF    = BL_OFF + NT * LDSP * 2;   // 30720

    __shared__ alignas(16) unsigned char smem[2 * BUF];

    const int t    = threadIdx.x;
    const int lane = t & 31;
    const int wave = t >> 5;
    const int nb   = blockIdx.x * NT;
    const int mb   = blockIdx.y * MT;
    const int wm   = (wave & 3) * 32;
    const int wn   = (wave >> 2) * 32;
    const unsigned lds0 = (unsigned)(uintptr_t)(void*)smem;  // low 32 bits = LDS byte addr

    v8f c00 = {}, c01 = {}, c10 = {}, c11 = {};

    auto issue = [&](int buf, int k0) {
        const unsigned base = lds0 + (unsigned)(buf * BUF);
        tdm_load_tile(base + AH_OFF, xh + (size_t)mb * DIN + k0, 8192u, MT);
        tdm_load_tile(base + AL_OFF, xl + (size_t)mb * DIN + k0, 8192u, MT);
        tdm_load_tile(base + BH_OFF, wh + (size_t)nb * DIN + k0, DOUT, NT);
        tdm_load_tile(base + BL_OFF, wl + (size_t)nb * DIN + k0, DOUT, NT);
    };

    if (wave == 0) issue(0, 0);

    const int nk = DIN / KT;
    for (int i = 0; i < nk; ++i) {
        const int cur = i & 1;
        if (wave == 0) {
            if (i + 1 < nk) {
                issue(cur ^ 1, (i + 1) * KT);            // prefetch next buffer via TDM
                __builtin_amdgcn_s_wait_tensorcnt(4);    // current buffer's 4 tiles done
            } else {
                __builtin_amdgcn_s_wait_tensorcnt(0);
            }
        }
        __syncthreads();

        const unsigned short* Ah = (const unsigned short*)(smem + cur * BUF + AH_OFF);
        const unsigned short* Al = (const unsigned short*)(smem + cur * BUF + AL_OFF);
        const unsigned short* Bh = (const unsigned short*)(smem + cur * BUF + BH_OFF);
        const unsigned short* Bl = (const unsigned short*)(smem + cur * BUF + BL_OFF);

        FragBF a0h, a0l, a1h, a1l, b0h, b0l, b1h, b1l;
        load_frag_A(Ah, wm,      a0h);  load_frag_A(Al, wm,      a0l);
        load_frag_A(Ah, wm + 16, a1h);  load_frag_A(Al, wm + 16, a1l);
        load_frag_B(Bh, wn,      b0h);  load_frag_B(Bl, wn,      b0l);
        load_frag_B(Bh, wn + 16, b1h);  load_frag_B(Bl, wn + 16, b1l);

        c00 = wmma_bf16(a0h, b0h, c00);
        c00 = wmma_bf16(a0h, b0l, c00);
        c00 = wmma_bf16(a0l, b0h, c00);

        c01 = wmma_bf16(a0h, b1h, c01);
        c01 = wmma_bf16(a0h, b1l, c01);
        c01 = wmma_bf16(a0l, b1h, c01);

        c10 = wmma_bf16(a1h, b0h, c10);
        c10 = wmma_bf16(a1h, b0l, c10);
        c10 = wmma_bf16(a1l, b0h, c10);

        c11 = wmma_bf16(a1h, b1h, c11);
        c11 = wmma_bf16(a1h, b1l, c11);
        c11 = wmma_bf16(a1l, b1h, c11);

        __syncthreads();
    }

    const int n0 = nb + wn + (lane & 15);
    const float bias0 = bias[n0];
    const float bias1 = bias[n0 + 16];
    const int moff = (lane >> 4) << 3;
#pragma unroll
    for (int r = 0; r < 8; ++r) {
        const int m0 = mb + wm + moff + r;
        out[(size_t)m0 * DOUT + n0]             = c00[r] + bias0;
        out[(size_t)m0 * DOUT + n0 + 16]        = c01[r] + bias1;
        out[(size_t)(m0 + 16) * DOUT + n0]      = c10[r] + bias0;
        out[(size_t)(m0 + 16) * DOUT + n0 + 16] = c11[r] + bias1;
    }
}

// ---- Kernel 5 (fallback path): round-1 GEMM with in-kernel split (small workspace) ----
__device__ __forceinline__ void split_store4(unsigned short* hi, unsigned short* lo,
                                             const float4& v) {
    uint2 ph, pl;
    split4(v, ph, pl);
    *(uint2*)hi = ph;
    *(uint2*)lo = pl;
}

__global__ __launch_bounds__(256) void gemm_wmma_kernel(const float* __restrict__ x,
                                                        const float* __restrict__ Weff,
                                                        const float* __restrict__ bias,
                                                        float* __restrict__ out) {
    __shared__ unsigned short Ah[MT * LDSP];
    __shared__ unsigned short Al[MT * LDSP];
    __shared__ unsigned short Bh[NT * LDSP];
    __shared__ unsigned short Bl[NT * LDSP];

    const int t    = threadIdx.x;
    const int lane = t & 31;
    const int wave = t >> 5;
    const int nb   = blockIdx.x * NT;
    const int mb   = blockIdx.y * MT;
    const int wm   = (wave & 3) * 32;
    const int wn   = (wave >> 2) * 32;

    v8f c00 = {}, c01 = {}, c10 = {}, c11 = {};

    for (int k0 = 0; k0 < DIN; k0 += KT) {
#pragma unroll
        for (int p = 0; p < 4; ++p) {
            int idx = p * 256 + t;
            int row = idx >> 3;
            int c4  = (idx & 7) << 2;
            const float4 v = *(const float4*)(x + (size_t)(mb + row) * DIN + k0 + c4);
            split_store4(&Ah[row * LDSP + c4], &Al[row * LDSP + c4], v);
        }
#pragma unroll
        for (int p = 0; p < 2; ++p) {
            int idx = p * 256 + t;
            int row = idx >> 3;
            int c4  = (idx & 7) << 2;
            const float4 v = *(const float4*)(Weff + (size_t)(nb + row) * DIN + k0 + c4);
            split_store4(&Bh[row * LDSP + c4], &Bl[row * LDSP + c4], v);
        }
        __syncthreads();

        FragBF a0h, a0l, a1h, a1l, b0h, b0l, b1h, b1l;
        load_frag_A(Ah, wm,      a0h);  load_frag_A(Al, wm,      a0l);
        load_frag_A(Ah, wm + 16, a1h);  load_frag_A(Al, wm + 16, a1l);
        load_frag_B(Bh, wn,      b0h);  load_frag_B(Bl, wn,      b0l);
        load_frag_B(Bh, wn + 16, b1h);  load_frag_B(Bl, wn + 16, b1l);

        c00 = wmma_bf16(a0h, b0h, c00);
        c00 = wmma_bf16(a0h, b0l, c00);
        c00 = wmma_bf16(a0l, b0h, c00);
        c01 = wmma_bf16(a0h, b1h, c01);
        c01 = wmma_bf16(a0h, b1l, c01);
        c01 = wmma_bf16(a0l, b1h, c01);
        c10 = wmma_bf16(a1h, b0h, c10);
        c10 = wmma_bf16(a1h, b0l, c10);
        c10 = wmma_bf16(a1l, b0h, c10);
        c11 = wmma_bf16(a1h, b1h, c11);
        c11 = wmma_bf16(a1h, b1l, c11);
        c11 = wmma_bf16(a1l, b1h, c11);

        __syncthreads();
    }

    const int n0 = nb + wn + (lane & 15);
    const float bias0 = bias[n0];
    const float bias1 = bias[n0 + 16];
    const int moff = (lane >> 4) << 3;
#pragma unroll
    for (int r = 0; r < 8; ++r) {
        const int m0 = mb + wm + moff + r;
        out[(size_t)m0 * DOUT + n0]             = c00[r] + bias0;
        out[(size_t)m0 * DOUT + n0 + 16]        = c01[r] + bias1;
        out[(size_t)(m0 + 16) * DOUT + n0]      = c10[r] + bias0;
        out[(size_t)(m0 + 16) * DOUT + n0 + 16] = c11[r] + bias1;
    }
}

extern "C" void kernel_launch(void* const* d_in, const int* in_sizes, int n_in,
                              void* d_out, int out_size, void* d_ws, size_t ws_size,
                              hipStream_t stream) {
    const float* x   = (const float*)d_in[0];
    const float* W   = (const float*)d_in[1];
    const float* b   = (const float*)d_in[2];
    const float* dv  = (const float*)d_in[3];
    const int*   dr  = (const int*)d_in[4];
    const int*   dc  = (const int*)d_in[5];
    float*       out = (float*)d_out;

    const int M = out_size / DOUT;                       // 8192
    const size_t wBytes   = (size_t)DOUT * DIN * 4;      // 64 MB  (W_eff f32)
    const size_t wSplit   = (size_t)DOUT * DIN * 2;      // 32 MB  (each bf16 plane)
    const size_t xSplit   = (size_t)M * DIN * 2;         // 64 MB  (each bf16 plane)
    const size_t wsNeeded = wBytes + 2 * wSplit + 2 * xSplit;  // 256 MB

    float* Weff = (float*)d_ws;

    // 1) W_eff = W ; 2) W_eff += scatter(delta)  (both paths need this)
    const int n4w = (DOUT * DIN) / 4;
    copy_w_kernel<<<(n4w + 255) / 256, 256, 0, stream>>>((const float4*)W, (float4*)Weff, n4w);
    const int nnz = in_sizes[3];
    scatter_kernel<<<(nnz + 255) / 256, 256, 0, stream>>>(dv, dr, dc, Weff, nnz);

    dim3 grid(DOUT / NT, M / MT);   // 64 x 64

    if (ws_size >= wsNeeded) {
        // Fast path: pre-split both operands into bf16 hi/lo planes, TDM-fed GEMM.
        unsigned short* wh = (unsigned short*)((char*)d_ws + wBytes);
        unsigned short* wl = (unsigned short*)((char*)d_ws + wBytes + wSplit);
        unsigned short* xh = (unsigned short*)((char*)d_ws + wBytes + 2 * wSplit);
        unsigned short* xl = (unsigned short*)((char*)d_ws + wBytes + 2 * wSplit + xSplit);

        split_plane_kernel<<<(n4w + 255) / 256, 256, 0, stream>>>(
            (const float4*)Weff, (uint2*)wh, (uint2*)wl, n4w);
        const int n4x = (int)(((size_t)M * DIN) / 4);
        split_plane_kernel<<<(n4x + 255) / 256, 256, 0, stream>>>(
            (const float4*)x, (uint2*)xh, (uint2*)xl, n4x);

        gemm_tdm_kernel<<<grid, 256, 0, stream>>>(xh, xl, wh, wl, b, out);
    } else {
        // Fallback: in-kernel split (round-1 path).
        gemm_wmma_kernel<<<grid, 256, 0, stream>>>(x, Weff, b, out);
    }
}
